// TransformerDecoderBlock_31018253812047
// MI455X (gfx1250) — matrix operand order repaired and checked by reading, not compiled, not run on hardware
//
#include <hip/hip_runtime.h>
#include <hip/hip_bf16.h>
#include <math.h>

#define DMODEL 2048
#define NH     16
#define DH     128
#define LAT    512
#define BATCH  2
#define SEQ    2048
#define NT     (BATCH*SEQ)   // 4096 token rows
#define EPS    1e-5f

typedef __bf16 bf16;
typedef __attribute__((ext_vector_type(16))) __bf16 v16bf;
typedef __attribute__((ext_vector_type(8)))  __bf16 v8bf;
typedef __attribute__((ext_vector_type(8)))  float  v8f;

__device__ __forceinline__ v8f wmma_bf16(v16bf a, v16bf b, v8f c) {
    return __builtin_amdgcn_wmma_f32_16x16x32_bf16(false, a, false, b, (short)0, c, false, false);
}
__device__ __forceinline__ v16bf cat16(v8bf lo, v8bf hi) {
    return __builtin_shufflevector(lo, hi, 0, 1, 2, 3, 4, 5, 6, 7, 8, 9, 10, 11, 12, 13, 14, 15);
}

// ---- CDNA5 async global->LDS copy (guarded; falls back to reg staging) ----
#if defined(__gfx1250__) && __has_builtin(__builtin_amdgcn_global_load_async_to_lds_b128)
#define HAVE_ASYNC_LDS 1
typedef int v4i_ __attribute__((vector_size(16)));
typedef __attribute__((address_space(1))) v4i_* as1v4p;   // global
typedef __attribute__((address_space(3))) v4i_* as3v4p;   // LDS
__device__ __forceinline__ void async_b128(const bf16* g, bf16* l) {
    __builtin_amdgcn_global_load_async_to_lds_b128(
        (as1v4p)(unsigned long long)(const void*)g,
        (as3v4p)(unsigned int)(unsigned long long)(void*)l, 0, 0);
}
#else
#define HAVE_ASYNC_LDS 0
#endif

// ---------------- f32 -> bf16 conversion ----------------
__global__ void f2bf_kernel(const float* __restrict__ in, bf16* __restrict__ out, size_t n) {
    size_t i = (size_t)blockIdx.x * blockDim.x + threadIdx.x;
    size_t stride = (size_t)gridDim.x * blockDim.x;
    for (; i < n; i += stride) out[i] = (bf16)in[i];
}

// ---------------- f32 [R,C] -> bf16 [C,R] (tiled transpose) ----------------
__global__ __launch_bounds__(256)
void f2bf_transpose_kernel(const float* __restrict__ in, bf16* __restrict__ out, int R, int C) {
    __shared__ float tile[32][33];
    int r0 = blockIdx.y * 32, c0 = blockIdx.x * 32;
    int tx = threadIdx.x & 31, ty = threadIdx.x >> 5;
#pragma unroll
    for (int i = ty; i < 32; i += 8) tile[i][tx] = in[(size_t)(r0 + i) * C + c0 + tx];
    __syncthreads();
#pragma unroll
    for (int i = ty; i < 32; i += 8) out[(size_t)(c0 + i) * R + r0 + tx] = (bf16)tile[tx][i];
}

// ---------------- V transpose: KV[:, D + h*DH + d] -> Vt[(b,h,d)][s] ----------------
__global__ __launch_bounds__(256)
void v_transpose_kernel(const bf16* __restrict__ KV, bf16* __restrict__ Vt) {
    int bh = blockIdx.z, b = bh / NH, h = bh % NH;
    int s0 = blockIdx.x * 32, d0 = blockIdx.y * 32;
    __shared__ bf16 t[32][33];
    int tx = threadIdx.x & 31, ty = threadIdx.x >> 5;
    const bf16* src = KV + ((size_t)b * SEQ) * (2 * DMODEL) + DMODEL + h * DH;
#pragma unroll
    for (int i = ty; i < 32; i += 8) t[i][tx] = src[(size_t)(s0 + i) * (2 * DMODEL) + d0 + tx];
    __syncthreads();
    bf16* dst = Vt + ((size_t)bh * DH) * SEQ;
#pragma unroll
    for (int i = ty; i < 32; i += 8) dst[(size_t)(d0 + i) * SEQ + s0 + tx] = t[tx][i];
}

// ---------------- LayerNorm (one block per row) -> bf16 ----------------
__global__ __launch_bounds__(256)
void ln_bf16_kernel(const float* __restrict__ in, const float* __restrict__ gamma,
                    const float* __restrict__ beta, bf16* __restrict__ out, int N) {
    int row = blockIdx.x;
    const float* x = in + (size_t)row * N;
    float s = 0.f, s2 = 0.f;
    for (int i = threadIdx.x; i < N; i += blockDim.x) { float v = x[i]; s += v; s2 += v * v; }
#pragma unroll
    for (int m = 1; m < 32; m <<= 1) { s += __shfl_xor(s, m, 32); s2 += __shfl_xor(s2, m, 32); }
    __shared__ float ws[8], ws2[8];
    int wave = threadIdx.x >> 5, lane = threadIdx.x & 31;
    if (lane == 0) { ws[wave] = s; ws2[wave] = s2; }
    __syncthreads();
    if (wave == 0) {
        float a = (lane < 8) ? ws[lane] : 0.f, a2 = (lane < 8) ? ws2[lane] : 0.f;
#pragma unroll
        for (int m = 1; m < 8; m <<= 1) { a += __shfl_xor(a, m, 32); a2 += __shfl_xor(a2, m, 32); }
        if (lane == 0) { ws[0] = a; ws2[0] = a2; }
    }
    __syncthreads();
    float mean = ws[0] / (float)N;
    float var  = ws2[0] / (float)N - mean * mean;
    float rstd = rsqrtf(var + EPS);
    for (int i = threadIdx.x; i < N; i += blockDim.x) {
        float v = (x[i] - mean) * rstd * gamma[i] + beta[i];
        out[(size_t)row * N + i] = (bf16)v;
    }
}

// ---------------- Tiled WMMA GEMM:  C[M,N] = A[M,K] @ W[N,K]^T ----------------
#define BM 128
#define BN 128
#define BK 32
#define LDT 40

template <int EPI>
__global__ __launch_bounds__(256)
void gemm_wmma_kernel(const bf16* __restrict__ A, const bf16* __restrict__ W,
                      const float* __restrict__ bias, const float* __restrict__ resid,
                      float* __restrict__ Cf, bf16* __restrict__ Cb,
                      int M, int N, int K) {
    __shared__ __attribute__((aligned(16))) bf16 As[2][BM * LDT];
    __shared__ __attribute__((aligned(16))) bf16 Bs[2][BN * LDT];
    int tid = threadIdx.x, lane = tid & 31, wave = tid >> 5;
    int m0 = blockIdx.y * BM, n0 = blockIdx.x * BN;
    int wm = (wave >> 1) * 32;
    int wn = (wave & 1) * 64;
    int lr = tid >> 1, lc = (tid & 1) * 16;
    const bf16* ga = A + (size_t)(m0 + lr) * K + lc;
    const bf16* gb = W + (size_t)(n0 + lr) * K + lc;

#if HAVE_ASYNC_LDS
    async_b128(ga, &As[0][lr * LDT + lc]); async_b128(ga + 8, &As[0][lr * LDT + lc + 8]);
    async_b128(gb, &Bs[0][lr * LDT + lc]); async_b128(gb + 8, &Bs[0][lr * LDT + lc + 8]);
    asm volatile("s_wait_asynccnt 0x0" ::: "memory");
#else
    {
        v8bf a0 = *(const v8bf*)ga, a1 = *(const v8bf*)(ga + 8);
        v8bf b0 = *(const v8bf*)gb, b1 = *(const v8bf*)(gb + 8);
        *(v8bf*)&As[0][lr * LDT + lc] = a0; *(v8bf*)&As[0][lr * LDT + lc + 8] = a1;
        *(v8bf*)&Bs[0][lr * LDT + lc] = b0; *(v8bf*)&Bs[0][lr * LDT + lc + 8] = b1;
    }
#endif
    __syncthreads();

    v8f acc[2][4] = {};
    int r = lane & 15, kg = lane >> 4;
    int nsteps = K / BK;
    for (int s = 0; s < nsteps; ++s) {
        int cur = s & 1;
        bool more = (s + 1 < nsteps);
#if HAVE_ASYNC_LDS
        if (more) {  // async copy next tile straight into the other LDS buffer
            const bf16* pa = ga + (size_t)(s + 1) * BK;
            const bf16* pb = gb + (size_t)(s + 1) * BK;
            bf16* la = &As[cur ^ 1][lr * LDT + lc];
            bf16* lb = &Bs[cur ^ 1][lr * LDT + lc];
            async_b128(pa, la); async_b128(pa + 8, la + 8);
            async_b128(pb, lb); async_b128(pb + 8, lb + 8);
        }
#else
        v8bf a0, a1, b0, b1;
        if (more) {
            const bf16* pa = ga + (size_t)(s + 1) * BK;
            const bf16* pb = gb + (size_t)(s + 1) * BK;
            a0 = *(const v8bf*)pa; a1 = *(const v8bf*)(pa + 8);
            b0 = *(const v8bf*)pb; b1 = *(const v8bf*)(pb + 8);
        }
#endif
        v16bf af[2], bfr[4];
#pragma unroll
        for (int tm = 0; tm < 2; ++tm) {
            const bf16* p = &As[cur][(wm + tm * 16 + r) * LDT + kg * 8];
            af[tm] = cat16(*(const v8bf*)p, *(const v8bf*)(p + 16));
        }
#pragma unroll
        for (int tn = 0; tn < 4; ++tn) {
            const bf16* p = &Bs[cur][(wn + tn * 16 + r) * LDT + kg * 8];
            bfr[tn] = cat16(*(const v8bf*)p, *(const v8bf*)(p + 16));
        }
#pragma unroll
        for (int tm = 0; tm < 2; ++tm)
#pragma unroll
            for (int tn = 0; tn < 4; ++tn)
                acc[tm][tn] = wmma_bf16(af[tm], bfr[tn], acc[tm][tn]);
#if HAVE_ASYNC_LDS
        if (more) asm volatile("s_wait_asynccnt 0x0" ::: "memory");
#else
        if (more) {
            *(v8bf*)&As[cur ^ 1][lr * LDT + lc] = a0; *(v8bf*)&As[cur ^ 1][lr * LDT + lc + 8] = a1;
            *(v8bf*)&Bs[cur ^ 1][lr * LDT + lc] = b0; *(v8bf*)&Bs[cur ^ 1][lr * LDT + lc + 8] = b1;
        }
#endif
        __syncthreads();
    }
    int hi = lane >> 4, nn = lane & 15;
#pragma unroll
    for (int tm = 0; tm < 2; ++tm)
#pragma unroll
        for (int tn = 0; tn < 4; ++tn)
#pragma unroll
            for (int i = 0; i < 8; ++i) {
                int row = m0 + wm + tm * 16 + hi * 8 + i;
                int col = n0 + wn + tn * 16 + nn;
                size_t idx = (size_t)row * N + col;
                float v = acc[tm][tn][i];
                if (EPI == 0)       Cf[idx] = v;
                else if (EPI == 1) { v += bias[col]; Cb[idx] = (bf16)(v > 0.f ? v : 0.f); }
                else if (EPI == 2)  Cf[idx] = v + resid[idx];
                else if (EPI == 3)  Cf[idx] = v + bias[col] + resid[idx];
                else                Cb[idx] = (bf16)v;
            }
}

// ---------------- Flash attention (causal), 64-wide kv chunks ----------------
template <bool MASKED>
__device__ __forceinline__ void attn_kv_step64(
    int kv0, int q0, int kg, int nn, int r,
    const bf16* __restrict__ Kbase, int kvstride, const bf16* __restrict__ Vtb,
    const v16bf (&qf)[4], v8f (&o)[8], float (&mi)[8], float (&li)[8],
    bf16 (*PsW)[72]) {
    const float scale = 0.08838834764831845f; // 1/sqrt(128)
    v8f st[4];
#pragma unroll
    for (int t = 0; t < 4; ++t) {  // S tiles: 16q x 16kv each
        v8f a = {};
#pragma unroll
        for (int kc = 0; kc < 4; ++kc) {
            const bf16* p = Kbase + (size_t)(kv0 + t * 16 + nn) * kvstride + kc * 32 + kg * 8;
            v16bf bfv = cat16(*(const v8bf*)p, *(const v8bf*)(p + 16));
            a = wmma_bf16(qf[kc], bfv, a);
        }
        st[t] = a;
    }
    // one online-softmax pass over 64 columns
#pragma unroll
    for (int i = 0; i < 8; ++i) {
        int row = q0 + kg * 8 + i;
        float v[4];
#pragma unroll
        for (int t = 0; t < 4; ++t) {
            v[t] = st[t][i] * scale;
            if (MASKED) { if (kv0 + t * 16 + nn > row) v[t] = -__builtin_inff(); }
        }
        float mx = fmaxf(fmaxf(v[0], v[1]), fmaxf(v[2], v[3]));
#pragma unroll
        for (int m = 1; m < 16; m <<= 1) mx = fmaxf(mx, __shfl_xor(mx, m, 32));
        float mnew = fmaxf(mi[i], mx);
        float rs = 0.f;
#pragma unroll
        for (int t = 0; t < 4; ++t) { float p = __expf(v[t] - mnew); st[t][i] = p; rs += p; }
#pragma unroll
        for (int m = 1; m < 16; m <<= 1) rs += __shfl_xor(rs, m, 32);
        float alpha = __expf(mi[i] - mnew);
        li[i] = li[i] * alpha + rs;
        mi[i] = mnew;
#pragma unroll
        for (int dc = 0; dc < 8; ++dc) o[dc][i] *= alpha;
    }
    // P (C-layout f32) -> LDS -> two A-fragments (16x64 bf16)
#pragma unroll
    for (int t = 0; t < 4; ++t)
#pragma unroll
        for (int i = 0; i < 8; ++i)
            PsW[kg * 8 + i][t * 16 + nn] = (bf16)st[t][i];
    asm volatile("s_wait_dscnt 0x0" ::: "memory");
    v16bf pf[2];
#pragma unroll
    for (int pc = 0; pc < 2; ++pc) {
        const bf16* p = &PsW[r][pc * 32 + kg * 8];
        pf[pc] = cat16(*(const v8bf*)p, *(const v8bf*)(p + 16));
    }
    asm volatile("" ::: "memory");
    // O += P @ V  (Vt: [d][kv] => contiguous kv fragments)
#pragma unroll
    for (int dc = 0; dc < 8; ++dc)
#pragma unroll
        for (int pc = 0; pc < 2; ++pc) {
            const bf16* p = Vtb + (size_t)(dc * 16 + nn) * SEQ + kv0 + pc * 32 + kg * 8;
            v16bf vf = cat16(*(const v8bf*)p, *(const v8bf*)(p + 16));
            o[dc] = wmma_bf16(pf[pc], vf, o[dc]);
        }
}

// grid: (BATCH*NH, SEQ/64); block: 128 (4 waves, 16 q-rows each)
__global__ __launch_bounds__(128)
void attn_flash_kernel(const bf16* __restrict__ Qb, const bf16* __restrict__ Kb,
                       const bf16* __restrict__ Vt, bf16* __restrict__ Ob, int kvstride) {
    int bh = blockIdx.x, b = bh / NH, h = bh % NH;
    int wave = threadIdx.x >> 5, lane = threadIdx.x & 31;
    int q0 = blockIdx.y * 64 + wave * 16;
    const bf16* Qbase = Qb + ((size_t)b * SEQ) * DMODEL + h * DH;
    const bf16* Kbase = Kb + ((size_t)b * SEQ) * kvstride + h * DH;
    const bf16* Vtb   = Vt + ((size_t)bh * DH) * SEQ;
    int r = lane & 15, kg = lane >> 4, nn = lane & 15;

    v16bf qf[4];
#pragma unroll
    for (int kc = 0; kc < 4; ++kc) {
        const bf16* p = Qbase + (size_t)(q0 + r) * DMODEL + kc * 32 + kg * 8;
        qf[kc] = cat16(*(const v8bf*)p, *(const v8bf*)(p + 16));
    }

    v8f o[8] = {};
    float mi[8], li[8];
#pragma unroll
    for (int i = 0; i < 8; ++i) { mi[i] = -__builtin_inff(); li[i] = 0.f; }

    __shared__ __attribute__((aligned(16))) bf16 Ps[4][16][72];

    int kbu   = (q0 >= 63) ? ((q0 - 63) / 64 + 1) : 0;   // fully-unmasked 64-chunks
    int kbmax = (q0 + 15) / 64;
    int kb = 0;
    for (; kb < kbu; ++kb)
        attn_kv_step64<false>(kb * 64, q0, kg, nn, r, Kbase, kvstride, Vtb, qf, o, mi, li, Ps[wave]);
    for (; kb <= kbmax; ++kb)
        attn_kv_step64<true>(kb * 64, q0, kg, nn, r, Kbase, kvstride, Vtb, qf, o, mi, li, Ps[wave]);

    // normalize & store
#pragma unroll
    for (int i = 0; i < 8; ++i) {
        float inv = 1.f / li[i];
        int row = q0 + kg * 8 + i;
#pragma unroll
        for (int dc = 0; dc < 8; ++dc)
            Ob[((size_t)b * SEQ + row) * DMODEL + h * DH + dc * 16 + nn] = (bf16)(o[dc][i] * inv);
    }
}

// ---------------- host orchestration ----------------
static inline size_t align_up(size_t x) { return (x + 255) & ~(size_t)255; }

extern "C" void kernel_launch(void* const* d_in, const int* in_sizes, int n_in,
                              void* d_out, int out_size, void* d_ws, size_t ws_size,
                              hipStream_t stream) {
    const float* x      = (const float*)d_in[0];
    const float* W_dq   = (const float*)d_in[1];
    const float* W_uq   = (const float*)d_in[2];
    const float* q_ln_g = (const float*)d_in[3];
    const float* q_ln_b = (const float*)d_in[4];
    const float* W_dkv  = (const float*)d_in[5];
    const float* W_ukv  = (const float*)d_in[6];
    const float* kv_ln_g= (const float*)d_in[7];
    const float* kv_ln_b= (const float*)d_in[8];
    const float* W_o    = (const float*)d_in[9];
    const float* n1_g   = (const float*)d_in[10];
    const float* n1_b   = (const float*)d_in[11];
    const float* n2_g   = (const float*)d_in[12];
    const float* n2_b   = (const float*)d_in[13];
    const float* fc1_w  = (const float*)d_in[14];
    const float* fc1_b  = (const float*)d_in[15];
    const float* fc2_w  = (const float*)d_in[16];
    const float* fc2_b  = (const float*)d_in[17];
    float* out = (float*)d_out;

    char* ws = (char*)d_ws;
    size_t off = 0;
    auto carve = [&](size_t bytes) { char* p = ws + off; off += align_up(bytes); return p; };

    bf16* wb_dqT  = (bf16*)carve((size_t)LAT * DMODEL * 2);
    bf16* wb_uqT  = (bf16*)carve((size_t)DMODEL * LAT * 2);
    bf16* wb_dkvT = (bf16*)carve((size_t)LAT * DMODEL * 2);
    bf16* wb_ukvT = (bf16*)carve((size_t)2 * DMODEL * LAT * 2);
    bf16* wb_o    = (bf16*)carve((size_t)DMODEL * DMODEL * 2);
    bf16* wb_fc1  = (bf16*)carve((size_t)4 * DMODEL * DMODEL * 2);
    bf16* wb_fc2  = (bf16*)carve((size_t)DMODEL * 4 * DMODEL * 2);
    bf16* normed = (bf16*)carve((size_t)NT * DMODEL * 2);
    float* cq_f  = (float*)carve((size_t)NT * LAT * 4);
    bf16* cq_bf  = (bf16*)carve((size_t)NT * LAT * 2);
    float* ckv_f = (float*)carve((size_t)NT * LAT * 4);
    bf16* ckv_bf = (bf16*)carve((size_t)NT * LAT * 2);
    bf16* Q_bf   = (bf16*)carve((size_t)NT * DMODEL * 2);
    bf16* KV_bf  = (bf16*)carve((size_t)NT * 2 * DMODEL * 2);
    bf16* Vt_bf  = (bf16*)carve((size_t)BATCH * NH * DH * SEQ * 2);
    bf16* O_bf   = (bf16*)carve((size_t)NT * DMODEL * 2);
    float* x2_f  = (float*)carve((size_t)NT * DMODEL * 4);
    bf16* h_bf   = (bf16*)carve((size_t)NT * DMODEL * 2);
    bf16* h1_bf  = (bf16*)carve((size_t)NT * 4 * DMODEL * 2);
    (void)ws_size; (void)n_in; (void)in_sizes; (void)out_size;

    f2bf_transpose_kernel<<<dim3(LAT / 32, DMODEL / 32), 256, 0, stream>>>(W_dq, wb_dqT, DMODEL, LAT);
    f2bf_transpose_kernel<<<dim3(DMODEL / 32, LAT / 32), 256, 0, stream>>>(W_uq, wb_uqT, LAT, DMODEL);
    f2bf_transpose_kernel<<<dim3(LAT / 32, DMODEL / 32), 256, 0, stream>>>(W_dkv, wb_dkvT, DMODEL, LAT);
    f2bf_transpose_kernel<<<dim3(2 * DMODEL / 32, LAT / 32), 256, 0, stream>>>(W_ukv, wb_ukvT, LAT, 2 * DMODEL);
    f2bf_kernel<<<2048, 256, 0, stream>>>(W_o,   wb_o,   (size_t)DMODEL * DMODEL);
    f2bf_kernel<<<2048, 256, 0, stream>>>(fc1_w, wb_fc1, (size_t)4 * DMODEL * DMODEL);
    f2bf_kernel<<<2048, 256, 0, stream>>>(fc2_w, wb_fc2, (size_t)DMODEL * 4 * DMODEL);

    // 1) normed = LN(x) -> bf16
    ln_bf16_kernel<<<NT, 256, 0, stream>>>(x, n1_g, n1_b, normed, DMODEL);

    // 2) cq/ckv projections + latent LN
    gemm_wmma_kernel<0><<<dim3(LAT / BN, NT / BM), 256, 0, stream>>>(
        normed, wb_dqT, nullptr, nullptr, cq_f, nullptr, NT, LAT, DMODEL);
    gemm_wmma_kernel<0><<<dim3(LAT / BN, NT / BM), 256, 0, stream>>>(
        normed, wb_dkvT, nullptr, nullptr, ckv_f, nullptr, NT, LAT, DMODEL);
    ln_bf16_kernel<<<NT, 256, 0, stream>>>(cq_f, q_ln_g, q_ln_b, cq_bf, LAT);
    ln_bf16_kernel<<<NT, 256, 0, stream>>>(ckv_f, kv_ln_g, kv_ln_b, ckv_bf, LAT);

    // 3) up-projections
    gemm_wmma_kernel<4><<<dim3(DMODEL / BN, NT / BM), 256, 0, stream>>>(
        cq_bf, wb_uqT, nullptr, nullptr, nullptr, Q_bf, NT, DMODEL, LAT);
    gemm_wmma_kernel<4><<<dim3(2 * DMODEL / BN, NT / BM), 256, 0, stream>>>(
        ckv_bf, wb_ukvT, nullptr, nullptr, nullptr, KV_bf, NT, 2 * DMODEL, LAT);

    // 3b) transpose V per (b,h)
    v_transpose_kernel<<<dim3(SEQ / 32, DH / 32, BATCH * NH), 256, 0, stream>>>(KV_bf, Vt_bf);

    // 4) causal flash attention
    attn_flash_kernel<<<dim3(BATCH * NH, SEQ / 64), 128, 0, stream>>>(
        Q_bf, KV_bf, Vt_bf, O_bf, 2 * DMODEL);

    // 5) x2 = x + O @ W_o^T
    gemm_wmma_kernel<2><<<dim3(DMODEL / BN, NT / BM), 256, 0, stream>>>(
        O_bf, wb_o, nullptr, x, x2_f, nullptr, NT, DMODEL, DMODEL);

    // 6) h = LN(x2) -> bf16
    ln_bf16_kernel<<<NT, 256, 0, stream>>>(x2_f, n2_g, n2_b, h_bf, DMODEL);

    // 7) h1 = relu(h @ fc1^T + b1) -> bf16
    gemm_wmma_kernel<1><<<dim3(4 * DMODEL / BN, NT / BM), 256, 0, stream>>>(
        h_bf, wb_fc1, fc1_b, nullptr, nullptr, h1_bf, NT, 4 * DMODEL, DMODEL);

    // 8) out = x2 + h1 @ fc2^T + b2
    gemm_wmma_kernel<3><<<dim3(DMODEL / BN, NT / BM), 256, 0, stream>>>(
        h1_bf, wb_fc2, fc2_b, x2_f, out, nullptr, NT, DMODEL, 4 * DMODEL);
}
